// StandardMultiHeadAttention_58265526338307
// MI455X (gfx1250) — matrix operand order repaired and checked by reading, not compiled, not run on hardware
//
#include <hip/hip_runtime.h>
#include <hip/hip_bf16.h>

// ---------------------------------------------------------------------------
// Types for CDNA5 WMMA (wave32): D(16x16 f32) = A(16x32 bf16) x B(32x16 bf16) + C
// ---------------------------------------------------------------------------
typedef __attribute__((ext_vector_type(16))) __bf16 v16bf;
typedef __attribute__((ext_vector_type(8)))  float  v8f;
typedef __attribute__((ext_vector_type(4)))  unsigned int u32x4;
typedef __attribute__((ext_vector_type(8)))  int    i32x8;
typedef __attribute__((ext_vector_type(4)))  int    i32x4;

struct U8 { uint4 a, b; };   // 32 bytes == one v16bf fragment

#if __has_builtin(__builtin_amdgcn_tensor_load_to_lds) && __has_builtin(__builtin_amdgcn_s_wait_tensorcnt)
#define HAVE_TDM 1
#else
#define HAVE_TDM 0
#endif

static __device__ __forceinline__ unsigned short f2bf_u(float f) {
  unsigned int u = __builtin_bit_cast(unsigned int, f);
  u += 0x7FFFu + ((u >> 16) & 1u);            // round-to-nearest-even
  return (unsigned short)(u >> 16);
}

static __device__ __forceinline__ v16bf frag(const uint4* p0, const uint4* p1) {
  U8 u{*p0, *p1};
  return __builtin_bit_cast(v16bf, u);
}

#if HAVE_TDM
typedef __attribute__((address_space(3))) unsigned short lds_us;

static __device__ __forceinline__ unsigned lds_off(void* p) {
  // workgroup-relative LDS byte address as used by DS / TDM
  return (unsigned)(unsigned long long)(lds_us*)p;
}

static __device__ __forceinline__ void tdm_issue(u32x4 g0, i32x8 g1) {
  i32x4 z4 = {};
#if __clang_major__ >= 23
  i32x8 z8 = {};
  __builtin_amdgcn_tensor_load_to_lds(g0, g1, z4, z4, z8, 0);
#else
  __builtin_amdgcn_tensor_load_to_lds(g0, g1, z4, z4, 0);
#endif
}

// D# group0: count=1, lds_addr, 57-bit global addr, type=2 ("image")
static __device__ __forceinline__ u32x4 tdm_g0(unsigned lds_addr, const void* gp) {
  unsigned long long ga = (unsigned long long)gp;
  u32x4 g0;
  g0[0] = 1u;                                           // count=1, user mode
  g0[1] = lds_addr;
  g0[2] = (unsigned)ga;
  g0[3] = (unsigned)((ga >> 32) & 0x1FFFFFFu) | (2u << 30);
  return g0;
}

// 1D contiguous copy: data_size=8B, tile_dim0 = n_qwords
static __device__ __forceinline__ void tdm_load_1d(unsigned lds_addr, const void* gp,
                                                   unsigned n_qwords) {
  i32x8 g1 = {};
  g1[0] = (3 << 16);                                    // data_size = 8 bytes
  g1[1] = (int)((n_qwords & 0xFFFFu) << 16);            // tensor_dim0 lo16
  g1[2] = (int)((n_qwords >> 16) & 0xFFFFu) | (1 << 16);// tensor_dim0 hi16 | tensor_dim1=1
  g1[3] = (int)(n_qwords << 16);                        // tile_dim0
  g1[4] = 1;                                            // tile_dim1 = 1
  g1[5] = (int)n_qwords;                                // tensor_dim0_stride
  tdm_issue(tdm_g0(lds_addr, gp), g1);
}

// 2D tile: data_size=2B (bf16), tile_w x tile_h, row stride in elements,
// LDS padding of pad_amount(code) dwords every pad_interval(code) dwords.
static __device__ __forceinline__ void tdm_load_2d(unsigned lds_addr, const void* gp,
                                                   unsigned tile_w, unsigned tile_h,
                                                   unsigned row_stride,
                                                   unsigned pad_int_code,
                                                   unsigned pad_amt_code) {
  const unsigned td1 = 1u << 20;                        // generous tensor_dim1
  i32x8 g1 = {};
  g1[0] = (1 << 16) | (1 << 20) |                       // data_size=2B, pad_enable
          (int)(pad_int_code << 22) | (int)(pad_amt_code << 25);
  g1[1] = (int)((row_stride & 0xFFFFu) << 16);          // tensor_dim0 lo16
  g1[2] = (int)((row_stride >> 16) & 0xFFFFu) |         // tensor_dim0 hi16
          (int)((td1 & 0xFFFFu) << 16);                 // tensor_dim1 lo16
  g1[3] = (int)((td1 >> 16) & 0xFFFFu) |                // tensor_dim1 hi16
          (int)(tile_w << 16);                          // tile_dim0
  g1[4] = (int)tile_h;                                  // tile_dim1 (tile_dim2=0)
  g1[5] = (int)row_stride;                              // tensor_dim0_stride lo32
  tdm_issue(tdm_g0(lds_addr, gp), g1);
}
#endif  // HAVE_TDM

// ---------------------------------------------------------------------------
// K1: hidden (B,C,N) fp32  ->  xb (B,N,C) bf16   (A-operand of QKV GEMM)
// ---------------------------------------------------------------------------
__global__ __launch_bounds__(256) void k_prep_x(const float* __restrict__ hid,
                                                unsigned short* __restrict__ xb) {
  int idx = blockIdx.x * 256 + threadIdx.x;   // c fastest -> coalesced store
  int c = idx & 1023;
  int n = (idx >> 10) & 1023;
  int b = idx >> 20;
  xb[idx] = f2bf_u(hid[((size_t)((b << 10) + c)) * 1024 + n]);
}

// K2: w (M,1024) fp32 -> wt (1024,M) bf16   (K-major B-operand)
__global__ __launch_bounds__(256) void k_prep_w(const float* __restrict__ w,
                                                unsigned short* __restrict__ wt, int M) {
  int idx = blockIdx.x * 256 + threadIdx.x;   // m fastest -> coalesced store
  int m = idx % M;
  int k = idx / M;
  wt[idx] = f2bf_u(w[(size_t)m * 1024 + k]);
}

// ---------------------------------------------------------------------------
// K3/K6: bf16 GEMM, C = A(M x 1024) x B(1024 x Nld).
// Block tile 64x128, 8 waves in 4x2: each wave 16 rows x 64 cols (4 C tiles).
// Double-buffered TDM pipeline: issue tile i+1, s_wait_tensorcnt 2 (TENSORcnt
// completes in order) so tile i is ready while i+1 streams in behind compute.
// TDM pad codes reproduce the LDS row strides 40 / 136 ushorts exactly.
// mode 0: QKV epilogue (q/k -> fp32 for RMSNorm, v -> bf16 per head)
// mode 1: plain fp32 store (proj)
// ---------------------------------------------------------------------------
__global__ __launch_bounds__(256) void k_gemm(
    const unsigned short* __restrict__ A,    // (M,1024) bf16 row-major
    const unsigned short* __restrict__ Bm,   // (1024,Nld) bf16 row-major
    int Nld, int mode,
    const float* __restrict__ bias,
    float* __restrict__ qk,
    unsigned short* __restrict__ vout,
    float* __restrict__ pout) {
  __shared__ __align__(16) unsigned short lA[2][64][40];    // 32 data + 8 pad per row
  __shared__ __align__(16) unsigned short lB[2][32][136];   // 128 data + 8 pad per row

  const int tid  = threadIdx.x;
  const int wave = tid >> 5, lane = tid & 31;
  const int hl = lane >> 4, l15 = lane & 15;
  const int wm = wave >> 1, wn = wave & 1;
  const int m0 = blockIdx.y * 64;
  const int n0 = blockIdx.x * 128;

  v8f acc[4];
  acc[0] = {}; acc[1] = {}; acc[2] = {}; acc[3] = {};

  const int arow = tid >> 2, acol = (tid & 3) * 8;       // 64x32 A tile, uint4 each
  const int brow = tid >> 3, bcol = (tid & 7) * 16;      // 32x128 B tile, 2x uint4

#if HAVE_TDM
  const unsigned ldsA[2] = { lds_off(&lA[0][0][0]), lds_off(&lA[1][0][0]) };
  const unsigned ldsB[2] = { lds_off(&lB[0][0][0]), lds_off(&lB[1][0][0]) };
  if (wave == 0) {   // prologue: tile 0 into buffer 0
    tdm_load_2d(ldsA[0], &A[(size_t)m0 * 1024], 32, 64, 1024, 3, 3);
    tdm_load_2d(ldsB[0], &Bm[n0], 128, 32, (unsigned)Nld, 5, 3);
  }
#endif

  for (int k0 = 0; k0 < 1024; k0 += 32) {
    const int ib = (k0 >> 5) & 1;
    __syncthreads();                 // buffer ib^1 free (last read one iter ago)
#if HAVE_TDM
    if (wave == 0) {
      if (k0 + 32 < 1024) {
        tdm_load_2d(ldsA[ib ^ 1], &A[(size_t)m0 * 1024 + k0 + 32], 32, 64, 1024, 3, 3);
        tdm_load_2d(ldsB[ib ^ 1], &Bm[(size_t)(k0 + 32) * Nld + n0], 128, 32,
                    (unsigned)Nld, 5, 3);
        __builtin_amdgcn_s_wait_tensorcnt(2);   // current tile's 2 loads done
      } else {
        __builtin_amdgcn_s_wait_tensorcnt(0);
      }
    }
#else
    *(uint4*)&lA[ib][arow][acol]     = *(const uint4*)&A[(size_t)(m0 + arow) * 1024 + k0 + acol];
    *(uint4*)&lB[ib][brow][bcol]     = *(const uint4*)&Bm[(size_t)(k0 + brow) * Nld + n0 + bcol];
    *(uint4*)&lB[ib][brow][bcol + 8] = *(const uint4*)&Bm[(size_t)(k0 + brow) * Nld + n0 + bcol + 8];
#endif
    if (k0 + 32 < 1024) {
      __builtin_prefetch(&A[(size_t)(m0 + arow) * 1024 + k0 + 32 + acol], 0, 0);
      __builtin_prefetch(&Bm[(size_t)(k0 + 32 + brow) * Nld + n0 + bcol], 0, 0);
    }
    __syncthreads();

    // A frag: lane = row M; VGPR0-3: K = hl*8 .. +7 ; VGPR4-7: K = 16+hl*8 .. +7
    const int mrow = wm * 16 + l15;
    v16bf af = frag((const uint4*)&lA[ib][mrow][hl * 8],
                    (const uint4*)&lA[ib][mrow][16 + hl * 8]);
#pragma unroll
    for (int nt = 0; nt < 4; ++nt) {
      // B frag: lane = column N (l15); elements: K = hl*8 .. +7, then 16+hl*8 .. +7
      const int cb = wn * 64 + nt * 16;
      unsigned short tb[16];
#pragma unroll
      for (int s = 0; s < 16; ++s) tb[s] = lB[ib][(s & 7) + hl * 8 + (s >> 3) * 16][cb + l15];
      v16bf bf = __builtin_bit_cast(v16bf, tb);
      acc[nt] = __builtin_amdgcn_wmma_f32_16x16x32_bf16(false, af, false, bf,
                                                        (short)0, acc[nt], false, false);
    }
  }

  // C layout: VGPR r holds row M = r + 8*hl, column N = l15
#pragma unroll
  for (int nt = 0; nt < 4; ++nt) {
    const int gc0 = n0 + wn * 64 + nt * 16;
    const int gcol = gc0 + l15;
#pragma unroll
    for (int r = 0; r < 8; ++r) {
      const int grow = m0 + wm * 16 + r + 8 * hl;
      float vv = acc[nt][r];
      if (mode == 0) {
        vv += bias[gcol];
        if (gc0 < 2048) {                               // q or k -> fp32 (B,N,2,1024)
          const int sel = gcol >> 10, cc = gcol & 1023;
          qk[((size_t)grow * 2 + sel) * 1024 + cc] = vv;
        } else {                                        // v -> bf16 (B,h,N,64)
          const int mm = gcol - 2048, h = mm >> 6, d = mm & 63;
          const int b = grow >> 10, n = grow & 1023;
          vout[(((size_t)(b * 16 + h) * 1024) + n) * 64 + d] = f2bf_u(vv);
        }
      } else {
        pout[(size_t)grow * Nld + gcol] = vv;
      }
    }
  }
}

// ---------------------------------------------------------------------------
// K4: 2D-RoPE + RMSNorm(inner=1024) for q and k; q gets hd^-0.5 folded in.
// ---------------------------------------------------------------------------
__global__ __launch_bounds__(256) void k_rope_norm(
    const float* __restrict__ qk,
    const float* __restrict__ qw, const float* __restrict__ kw,
    unsigned short* __restrict__ qb, unsigned short* __restrict__ kb) {
  __shared__ float redq[8], redk[8];
  const int g = blockIdx.x;
  const int b = g >> 10, n = g & 1023;
  const float hp = (float)(n >> 5), wp = (float)(n & 31);
  const int t = threadIdx.x;
  const int i0 = t * 4;

  float4 q4 = *(const float4*)&qk[((size_t)g * 2 + 0) * 1024 + i0];
  float4 k4 = *(const float4*)&qk[((size_t)g * 2 + 1) * 1024 + i0];

  const int d0 = i0 & 63;
  const int j0 = d0 >> 1, j1 = j0 + 1;
  // freq(i) = 10000^(-i/16) = exp(-i * ln(10000)/16)
  float a0 = ((j0 < 16) ? hp : wp) * __expf(-0.5756462733f * (float)(j0 & 15));
  float a1 = ((j1 < 16) ? hp : wp) * __expf(-0.5756462733f * (float)(j1 & 15));
  float c0 = __cosf(a0), s0 = __sinf(a0), c1 = __cosf(a1), s1 = __sinf(a1);

  float rq[4], rk[4];
  rq[0] = q4.x * c0 - q4.y * s0; rq[1] = q4.x * s0 + q4.y * c0;
  rq[2] = q4.z * c1 - q4.w * s1; rq[3] = q4.z * s1 + q4.w * c1;
  rk[0] = k4.x * c0 - k4.y * s0; rk[1] = k4.x * s0 + k4.y * c0;
  rk[2] = k4.z * c1 - k4.w * s1; rk[3] = k4.z * s1 + k4.w * c1;

  float ssq = rq[0]*rq[0] + rq[1]*rq[1] + rq[2]*rq[2] + rq[3]*rq[3];
  float ssk = rk[0]*rk[0] + rk[1]*rk[1] + rk[2]*rk[2] + rk[3]*rk[3];
#pragma unroll
  for (int m = 1; m < 32; m <<= 1) { ssq += __shfl_xor(ssq, m); ssk += __shfl_xor(ssk, m); }
  if ((t & 31) == 0) { redq[t >> 5] = ssq; redk[t >> 5] = ssk; }
  __syncthreads();
  float tq = 0.f, tk = 0.f;
#pragma unroll
  for (int wv = 0; wv < 8; ++wv) { tq += redq[wv]; tk += redk[wv]; }
  const float rmq = rsqrtf(tq * (1.f / 1024.f) + 1e-6f) * 0.125f;  // fold hd^-0.5
  const float rmk = rsqrtf(tk * (1.f / 1024.f) + 1e-6f);

#pragma unroll
  for (int e = 0; e < 4; ++e) {
    const int i = i0 + e, h = i >> 6, d = i & 63;
    const size_t o = (((size_t)(b * 16 + h) * 1024) + n) * 64 + d;
    qb[o] = f2bf_u(rq[e] * rmq * qw[i]);
    kb[o] = f2bf_u(rk[e] * rmk * kw[i]);
  }
}

// ---------------------------------------------------------------------------
// K5: flash attention per (qblock=128 rows, head, batch); 8 waves x 16 q rows.
// K/V key-blocks are contiguous 4KB -> 1D TDM loads, double-buffered so the
// next block streams in while the current one is consumed by WMMA.
// Per 32-key step: 4 WMMA for S = Q K^T, online softmax (half-wave shuffles),
// P transposed via per-wave LDS, 4 WMMA for O += P V.
// ---------------------------------------------------------------------------
__global__ __launch_bounds__(256) void k_attn(
    const unsigned short* __restrict__ qm,
    const unsigned short* __restrict__ km,
    const unsigned short* __restrict__ vm,
    unsigned short* __restrict__ ctx) {
  __shared__ __align__(16) unsigned short Kraw[2][2048];   // 32 k-rows x 64 d
  __shared__ __align__(16) unsigned short Vraw[2][2048];   // 32 k-rows x 64 d
  __shared__ __align__(16) unsigned short Kt[64][40];      // K^T: [d][k]
  __shared__ __align__(16) unsigned short Pl[8][16][40];   // per-wave P staging

  const int tid = threadIdx.x;
  const int w = tid >> 5, lane = tid & 31, hl = lane >> 4, l15 = lane & 15;
  const int qb = blockIdx.x, h = blockIdx.y, b = blockIdx.z;
  const size_t bh = (size_t)(b * 16 + h) * 1024;

  // Q fragments for this wave's 16 rows (d=0..31 and d=32..63)
  const int qrow = qb * 128 + w * 16 + l15;
  const unsigned short* qp = &qm[(bh + qrow) * 64];
  v16bf aq0 = frag((const uint4*)&qp[hl * 8],      (const uint4*)&qp[16 + hl * 8]);
  v16bf aq1 = frag((const uint4*)&qp[32 + hl * 8], (const uint4*)&qp[48 + hl * 8]);

  v8f O[4]; O[0] = {}; O[1] = {}; O[2] = {}; O[3] = {};
  float mrow[8], lrow[8];
#pragma unroll
  for (int r = 0; r < 8; ++r) { mrow[r] = -3.0e38f; lrow[r] = 0.f; }

  const int krw = tid >> 3, kdb = (tid & 7) * 8;
#if HAVE_TDM
  const unsigned ldsK[2] = { lds_off(Kraw[0]), lds_off(Kraw[1]) };
  const unsigned ldsV[2] = { lds_off(Vraw[0]), lds_off(Vraw[1]) };
  if (w == 0) {   // prologue: key-block 0 into buffer 0
    tdm_load_1d(ldsK[0], &km[bh * 64], 512);
    tdm_load_1d(ldsV[0], &vm[bh * 64], 512);
  }
#endif

  for (int kt = 0; kt < 1024; kt += 32) {
    const int ib = (kt >> 5) & 1;
    __syncthreads();                 // buffer ib^1 free (last read one iter ago)
#if HAVE_TDM
    if (w == 0) {
      if (kt + 32 < 1024) {
        tdm_load_1d(ldsK[ib ^ 1], &km[(bh + kt + 32) * 64], 512);
        tdm_load_1d(ldsV[ib ^ 1], &vm[(bh + kt + 32) * 64], 512);
        __builtin_amdgcn_s_wait_tensorcnt(2);   // current K/V done, next in flight
      } else {
        __builtin_amdgcn_s_wait_tensorcnt(0);
      }
    }
#else
    *(uint4*)&Kraw[ib][krw * 64 + kdb] = *(const uint4*)&km[(bh + kt + krw) * 64 + kdb];
    *(uint4*)&Vraw[ib][krw * 64 + kdb] = *(const uint4*)&vm[(bh + kt + krw) * 64 + kdb];
#endif
    __syncthreads();

    {   // transpose K tile in LDS: Kt[d][krow]
      uint4 kv = *(const uint4*)&Kraw[ib][krw * 64 + kdb];
      const unsigned short* ks = (const unsigned short*)&kv;
#pragma unroll
      for (int i = 0; i < 8; ++i) Kt[kdb + i][krw] = ks[i];
    }
    __syncthreads();

    // S tiles: k columns [0..15] and [16..31] of this key block
    v16bf bk0a = frag((const uint4*)&Kraw[ib][l15 * 64 + hl * 8],             (const uint4*)&Kraw[ib][l15 * 64 + 16 + hl * 8]);
    v16bf bk1a = frag((const uint4*)&Kraw[ib][l15 * 64 + 32 + hl * 8],        (const uint4*)&Kraw[ib][l15 * 64 + 48 + hl * 8]);
    v16bf bk0b = frag((const uint4*)&Kraw[ib][(16 + l15) * 64 + hl * 8],      (const uint4*)&Kraw[ib][(16 + l15) * 64 + 16 + hl * 8]);
    v16bf bk1b = frag((const uint4*)&Kraw[ib][(16 + l15) * 64 + 32 + hl * 8], (const uint4*)&Kraw[ib][(16 + l15) * 64 + 48 + hl * 8]);
    v8f s0 = {}, s1 = {};
    s0 = __builtin_amdgcn_wmma_f32_16x16x32_bf16(false, aq0, false, bk0a, (short)0, s0, false, false);
    s0 = __builtin_amdgcn_wmma_f32_16x16x32_bf16(false, aq1, false, bk1a, (short)0, s0, false, false);
    s1 = __builtin_amdgcn_wmma_f32_16x16x32_bf16(false, aq0, false, bk0b, (short)0, s1, false, false);
    s1 = __builtin_amdgcn_wmma_f32_16x16x32_bf16(false, aq1, false, bk1b, (short)0, s1, false, false);

    // online softmax: row = VGPR slot, 16 columns live in the half-wave lanes
    float alpha[8];
#pragma unroll
    for (int r = 0; r < 8; ++r) {
      float mx = fmaxf(s0[r], s1[r]);
#pragma unroll
      for (int msk = 1; msk < 16; msk <<= 1) mx = fmaxf(mx, __shfl_xor(mx, msk));
      const float nm = fmaxf(mrow[r], mx);
      alpha[r] = __expf(mrow[r] - nm);
      const float p0 = __expf(s0[r] - nm);
      const float p1 = __expf(s1[r] - nm);
      float rs = p0 + p1;
#pragma unroll
      for (int msk = 1; msk < 16; msk <<= 1) rs += __shfl_xor(rs, msk);
      lrow[r] = lrow[r] * alpha[r] + rs;
      mrow[r] = nm;
      // transpose P from C layout to A layout through LDS
      Pl[w][r + 8 * hl][l15]      = f2bf_u(p0);
      Pl[w][r + 8 * hl][16 + l15] = f2bf_u(p1);
    }
    asm volatile("s_wait_dscnt 0" ::: "memory");

#pragma unroll
    for (int nt = 0; nt < 4; ++nt)
#pragma unroll
      for (int r = 0; r < 8; ++r) O[nt][r] *= alpha[r];

    v16bf pa = frag((const uint4*)&Pl[w][l15][hl * 8], (const uint4*)&Pl[w][l15][16 + hl * 8]);
#pragma unroll
    for (int nt = 0; nt < 4; ++nt) {   // V fragment from LDS (B layout: lane = d column, elements = key)
      const unsigned short* vp = &Vraw[ib][nt * 16 + l15];
      unsigned short tv[16];
#pragma unroll
      for (int s = 0; s < 16; ++s) tv[s] = vp[((s & 7) + hl * 8 + (s >> 3) * 16) * 64];
      v16bf bv = __builtin_bit_cast(v16bf, tv);
      O[nt] = __builtin_amdgcn_wmma_f32_16x16x32_bf16(false, pa, false, bv,
                                                      (short)0, O[nt], false, false);
    }
  }

  float linv[8];
#pragma unroll
  for (int r = 0; r < 8; ++r) linv[r] = 1.f / lrow[r];
#pragma unroll
  for (int nt = 0; nt < 4; ++nt)
#pragma unroll
    for (int r = 0; r < 8; ++r) {
      const int row = qb * 128 + w * 16 + r + 8 * hl;
      const int d = nt * 16 + l15;
      ctx[((size_t)(b * 1024 + row)) * 1024 + h * 64 + d] = f2bf_u(O[nt][r] * linv[r]);
    }
}

// ---------------------------------------------------------------------------
// K7: RMSNorm over C + residual + transpose back to (B,C,H,W)
// ---------------------------------------------------------------------------
__global__ __launch_bounds__(256) void k_final(
    const float* __restrict__ proj, const float* __restrict__ w,
    const float* __restrict__ hid, float* __restrict__ out) {
  __shared__ float red[8];
  const int g = blockIdx.x, b = g >> 10, n = g & 1023;
  const int t = threadIdx.x, i0 = t * 4;
  float4 p4 = *(const float4*)&proj[(size_t)g * 1024 + i0];
  float ss = p4.x * p4.x + p4.y * p4.y + p4.z * p4.z + p4.w * p4.w;
#pragma unroll
  for (int m = 1; m < 32; m <<= 1) ss += __shfl_xor(ss, m);
  if ((t & 31) == 0) red[t >> 5] = ss;
  __syncthreads();
  float tt = 0.f;
#pragma unroll
  for (int wv = 0; wv < 8; ++wv) tt += red[wv];
  const float rm = rsqrtf(tt * (1.f / 1024.f) + 1e-6f);
  const float pv[4] = {p4.x, p4.y, p4.z, p4.w};
#pragma unroll
  for (int e = 0; e < 4; ++e) {
    const int c = i0 + e;
    const size_t o = ((size_t)(b * 1024 + c)) * 1024 + n;
    out[o] = pv[e] * rm * w[c] + hid[o];
  }
}

// ---------------------------------------------------------------------------
extern "C" void kernel_launch(void* const* d_in, const int* in_sizes, int n_in,
                              void* d_out, int out_size, void* d_ws, size_t ws_size,
                              hipStream_t stream) {
  const float* hid  = (const float*)d_in[0];
  const float* wqkv = (const float*)d_in[1];
  const float* bqkv = (const float*)d_in[2];
  const float* qw   = (const float*)d_in[3];
  const float* kw   = (const float*)d_in[4];
  const float* wprj = (const float*)d_in[5];
  const float* now  = (const float*)d_in[6];
  float* out = (float*)d_out;
  (void)in_sizes; (void)n_in; (void)out_size; (void)ws_size;

  char* ws = (char*)d_ws;
  size_t off = 0;
  auto alloc = [&](size_t bytes) {
    char* p = ws + off;
    off = (off + bytes + 255) & ~(size_t)255;
    return p;
  };
  unsigned short* xb  = (unsigned short*)alloc(4ull * 1024 * 1024 * 2);        // x bf16 (B,N,C)
  unsigned short* wqt = (unsigned short*)alloc(3072ull * 1024 * 2);            // w_qkv^T bf16
  unsigned short* wpt = (unsigned short*)alloc(1024ull * 1024 * 2);            // w_proj^T bf16
  float*          qkf = (float*)alloc(4ull * 1024 * 2 * 1024 * 4);             // q,k fp32 pre-norm
  unsigned short* vb  = (unsigned short*)alloc(4ull * 16 * 1024 * 64 * 2);     // v bf16 (B,h,N,d)
  unsigned short* qbm = (unsigned short*)alloc(4ull * 16 * 1024 * 64 * 2);     // q bf16
  unsigned short* kbm = (unsigned short*)alloc(4ull * 16 * 1024 * 64 * 2);     // k bf16
  unsigned short* cx  = (unsigned short*)alloc(4ull * 1024 * 1024 * 2);        // ctx bf16 (B,N,inner)
  float*          pj  = (float*)alloc(4ull * 1024 * 1024 * 4);                 // proj fp32

  k_prep_x<<<4 * 1024 * 1024 / 256, 256, 0, stream>>>(hid, xb);
  k_prep_w<<<3072 * 1024 / 256, 256, 0, stream>>>(wqkv, wqt, 3072);
  k_prep_w<<<1024 * 1024 / 256, 256, 0, stream>>>(wprj, wpt, 1024);
  k_gemm<<<dim3(24, 64), 256, 0, stream>>>(xb, wqt, 3072, 0, bqkv, qkf, vb, nullptr);
  k_rope_norm<<<4096, 256, 0, stream>>>(qkf, qw, kw, qbm, kbm);
  k_attn<<<dim3(8, 16, 4), 256, 0, stream>>>(qbm, kbm, vb, cx);
  k_gemm<<<dim3(8, 64), 256, 0, stream>>>(cx, wpt, 1024, 1, nullptr, nullptr, nullptr, pj);
  k_final<<<4096, 256, 0, stream>>>(pj, now, hid, out);
}
